// GraphToGrid_53455162966688
// MI455X (gfx1250) — matrix operand order repaired and checked by reading, not compiled, not run on hardware
//
#include <hip/hip_runtime.h>
#include <hip/hip_bf16.h>

typedef __attribute__((ext_vector_type(16))) _Float16 v16h;
typedef __attribute__((ext_vector_type(8)))  _Float16 v8h;
typedef __attribute__((ext_vector_type(2)))  _Float16 v2h;
typedef __attribute__((ext_vector_type(8)))  float    v8f;

#define GRID_G 96
#define NNODES 4096
#define DFEAT  128
#define NBATCH 8
#define NPER   512               // nodes per batch
#define NPIX   (GRID_G*GRID_G)   // 9216

// ---------------------------------------------------------------------------
// Kernel 1: per-node separable Gaussian rows + normalizer.
// wy'[n][y] = exp(-(y-py)^2/(2s^2)) / (Sy*Sx + 1e-8),  wx[n][x] = exp(-(x-px)^2/(2s^2))
// ---------------------------------------------------------------------------
__global__ __launch_bounds__(96) void splat_precompute(
    const float* __restrict__ pos, const float* __restrict__ log_sigma,
    float* __restrict__ wyp, float* __restrict__ wxo)
{
    __shared__ float ey[GRID_G];
    __shared__ float ex[GRID_G];
    __shared__ float ssum[2];

    const int n = blockIdx.x;
    const int t = threadIdx.x;

    const float sigma = __expf(log_sigma[0]);
    const float inv2  = 1.0f / (2.0f * sigma * sigma);
    const float py = pos[2 * n + 0];
    const float px = pos[2 * n + 1];

    const float c  = (float)t;
    const float dy = c - py;
    const float dx = c - px;
    const float vy = __expf(-(dy * dy) * inv2);
    const float vx = __expf(-(dx * dx) * inv2);
    ey[t] = vy;
    ex[t] = vx;
    __syncthreads();
    if (t == 0) {
        float sy = 0.f, sx = 0.f;
        for (int i = 0; i < GRID_G; ++i) { sy += ey[i]; sx += ex[i]; }
        ssum[0] = sy; ssum[1] = sx;
    }
    __syncthreads();
    const float inv = 1.0f / (ssum[0] * ssum[1] + 1e-8f);
    wyp[n * GRID_G + t] = vy * inv;
    wxo[n * GRID_G + t] = vx;
}

// ---------------------------------------------------------------------------
// Kernel 2: transpose + pack features to f16 A[b][d][n_local]
// ---------------------------------------------------------------------------
__global__ __launch_bounds__(256) void splat_pack(
    const float* __restrict__ f, _Float16* __restrict__ A)
{
    const int idx = blockIdx.x * blockDim.x + threadIdx.x;
    if (idx >= NNODES * DFEAT) return;
    const int n  = idx >> 7;
    const int d  = idx & (DFEAT - 1);
    const int b  = n >> 9;
    const int nl = n & (NPER - 1);
    A[(((b * DFEAT) + d) << 9) | nl] = (_Float16)f[idx];
}

// ---------------------------------------------------------------------------
// Kernel 3: double-buffered, software-pipelined WMMA GEMM splat.
// Per workgroup: batch b, 64-pixel tile.  C(128 x 64) += A(128 x 512) * W(512 x 64)
// ---------------------------------------------------------------------------
__global__ __launch_bounds__(256) void splat_gemm(
    const _Float16* __restrict__ Ag, const float* __restrict__ wyp,
    const float* __restrict__ wxo, float* __restrict__ out)
{
    __shared__ _Float16 Alds[2][DFEAT][32];   // 2 x 8 KB
    __shared__ _Float16 Blds[2][64][32];      // 2 x 4 KB

    const int b  = blockIdx.y;
    const int p0 = blockIdx.x * 64;
    const int t  = threadIdx.x;
    const int lane = t & 31;
    const int wv   = t >> 5;
    const int l15  = lane & 15;
    const int half = lane >> 4;

    const _Float16* Abase = Ag + (size_t)b * DFEAT * NPER;
    const int nbase = b * NPER;

    // --- hoisted per-thread staging geometry (loop-invariant) ---
    int arow[4], aoff[4], bcol[4], bkl[4], by[4], bx[4];
    #pragma unroll
    for (int j = 0; j < 4; ++j) {
        const int idx = j * 256 + t;          // 0..1023
        arow[j] = idx >> 3;                   // A row 0..127
        aoff[j] = (idx & 7) * 4;              // A k-offset (halves)
        bcol[j] = idx >> 4;                   // B col 0..63
        bkl[j]  = (idx & 15) * 2;             // B k-offset 0..30
        const int p = p0 + bcol[j];
        by[j] = p / GRID_G;
        bx[j] = p - by[j] * GRID_G;
    }

    unsigned long long areg[4];
    float w0r[4], w1r[4];

    auto issue_loads = [&](int kc) {
        #pragma unroll
        for (int j = 0; j < 4; ++j)
            areg[j] = *(const unsigned long long*)(Abase + arow[j] * NPER + kc + aoff[j]);
        #pragma unroll
        for (int j = 0; j < 4; ++j) {
            const int n = nbase + kc + bkl[j];
            w0r[j] = wyp[n * GRID_G + by[j]]       * wxo[n * GRID_G + bx[j]];
            w1r[j] = wyp[(n + 1) * GRID_G + by[j]] * wxo[(n + 1) * GRID_G + bx[j]];
        }
    };
    auto commit = [&](int buf) {
        #pragma unroll
        for (int j = 0; j < 4; ++j)
            *(unsigned long long*)&Alds[buf][arow[j]][aoff[j]] = areg[j];
        #pragma unroll
        for (int j = 0; j < 4; ++j) {
            v2h h; h[0] = (_Float16)w0r[j]; h[1] = (_Float16)w1r[j];
            *(v2h*)&Blds[buf][bcol[j]][bkl[j]] = h;
        }
    };

    v8f acc0 = {}, acc1 = {}, acc2 = {}, acc3 = {};

    // prologue: stage chunk 0
    issue_loads(0);
    commit(0);
    __syncthreads();

    for (int kc = 0, buf = 0; kc < NPER; kc += 32, buf ^= 1) {
        const bool has_next = (kc + 32) < NPER;
        if (has_next) issue_loads(kc + 32);   // global loads fly under the WMMAs

        // A fragment: lanes 0-15: M=l15, K=0..15; lanes 16-31: K=16..31
        union { v16h v; v8h h[2]; } au;
        const _Float16* ap = &Alds[buf][wv * 16 + l15][half * 16];
        au.h[0] = *(const v8h*)ap;
        au.h[1] = *(const v8h*)(ap + 8);

        // B fragments: lanes 0-15: N=l15, K=0..15; lanes 16-31: K=16..31
        const v16h bf0 = *(const v16h*)&Blds[buf][ 0 + l15][half * 16];
        const v16h bf1 = *(const v16h*)&Blds[buf][16 + l15][half * 16];
        const v16h bf2 = *(const v16h*)&Blds[buf][32 + l15][half * 16];
        const v16h bf3 = *(const v16h*)&Blds[buf][48 + l15][half * 16];

        acc0 = __builtin_amdgcn_wmma_f32_16x16x32_f16(false, au.v, false, bf0, (short)0, acc0, false, false);
        acc1 = __builtin_amdgcn_wmma_f32_16x16x32_f16(false, au.v, false, bf1, (short)0, acc1, false, false);
        acc2 = __builtin_amdgcn_wmma_f32_16x16x32_f16(false, au.v, false, bf2, (short)0, acc2, false, false);
        acc3 = __builtin_amdgcn_wmma_f32_16x16x32_f16(false, au.v, false, bf3, (short)0, acc3, false, false);

        if (has_next) {
            commit(buf ^ 1);
            __syncthreads();
        }
    }

    // --- writeback: lane holds col n=l15, rows m = wv*16 + half*8 + i ---
    const int mrow = wv * 16 + half * 8;
    float* obase = out + ((size_t)(b * DFEAT + mrow)) * NPIX + p0 + l15;
    v8f accs[4] = {acc0, acc1, acc2, acc3};
    #pragma unroll
    for (int c = 0; c < 4; ++c) {
        float* o = obase + c * 16;
        #pragma unroll
        for (int i = 0; i < 8; ++i)
            __builtin_nontemporal_store(accs[c][i], o + (size_t)i * NPIX);
    }
}

// ---------------------------------------------------------------------------
extern "C" void kernel_launch(void* const* d_in, const int* in_sizes, int n_in,
                              void* d_out, int out_size, void* d_ws, size_t ws_size,
                              hipStream_t stream) {
    const float* node_features = (const float*)d_in[0];   // (N, D) f32
    const float* node_pos      = (const float*)d_in[1];   // (N, 2) f32
    // d_in[2] = batch_idx (sorted, equal counts -> implicit layout)
    // d_in[3] = batch_size (fixed 8)
    const float* log_sigma     = (const float*)d_in[4];   // scalar f32

    float*    wyp = (float*)d_ws;                         // N*96 f32
    float*    wxo = wyp + NNODES * GRID_G;                // N*96 f32
    _Float16* Af  = (_Float16*)(wxo + NNODES * GRID_G);   // B*D*NPER f16

    splat_precompute<<<NNODES, GRID_G, 0, stream>>>(node_pos, log_sigma, wyp, wxo);
    splat_pack<<<(NNODES * DFEAT) / 256, 256, 0, stream>>>(node_features, Af);

    dim3 grid(NPIX / 64, NBATCH);   // 144 x 8 workgroups
    splat_gemm<<<grid, 256, 0, stream>>>(Af, wyp, wxo, (float*)d_out);
}
